// GRUDDecoder_45037027065875
// MI455X (gfx1250) — matrix operand order repaired and checked by reading, not compiled, not run on hardware
//
#include <hip/hip_runtime.h>

// ---------------------------------------------------------------------------
// GRU-D decode for MI455X (gfx1250), wave32 + WMMA bf16 + async LDS staging.
//
// Pipeline:
//   K0 pack_b        : fp32 weights -> bf16 packed WMMA-B fragments (one-time)
//   K1 grud_scan     : per-(b,h) elementwise scans -> delta/x_hat/mask (bf16)
//   K2 gemm_gamma    : gamma_h = exp(-relu(delta @ W_gh + b_gh))      [WMMA]
//   K3 gemm_pre      : pre = x_hat @ Wx + mask @ Wm + b   (fp32 out)  [WMMA]
//   K4 grud_recur    : sequential h-recurrence, 8 WGs x 16 batch rows [WMMA]
//   K5 gemm_out      : out = h_seq @ W_out + b_out        (fp32 out)  [WMMA]
//
// K2/K3/K5 stage their block-shared A-tile (16x512 bf16) into LDS once via
// GLOBAL_LOAD_ASYNC_TO_LDS_B128 (ASYNCcnt / s_wait_asynccnt), removing the
// 8x per-wave redundant A loads.
// ---------------------------------------------------------------------------

typedef __attribute__((ext_vector_type(16))) __bf16       v16bf;
typedef __attribute__((ext_vector_type(8)))  float        v8f;
typedef __attribute__((ext_vector_type(4)))  unsigned int u32x4;

static constexpr int Bn   = 128;
static constexpr int Ln   = 256;
static constexpr int Hn   = 512;
static constexpr int H3n  = 1536;
static constexpr int OUTn = 256;
static constexpr int BLn  = Bn * Ln;
static constexpr int HPAD = 520;          // LDS row stride (elems), bank spread

union FragU { v16bf v; u32x4 q[2]; };

__device__ __forceinline__ __bf16 f2bf(float x) { return (__bf16)x; }

#define WMMA_BF16(a, b, c) \
  __builtin_amdgcn_wmma_f32_16x16x32_bf16(false, (a), false, (b), (short)0, (c), false, false)

// A fragment (16x32 bf16, M x K) from row-major bf16: lane = row (lane&15),
// two contiguous 8-elem K runs at kbase + half*8 and kbase + 16 + half*8.
__device__ __forceinline__ v16bf load_a_rm(const __bf16* __restrict__ A, int lda,
                                           int mbase, int kbase, int lane) {
  int row = mbase + (lane & 15);
  int kh  = (lane >> 4) * 8;
  const __bf16* p = A + (size_t)row * lda + kbase + kh;
  FragU f;
  f.q[0] = *(const u32x4*)(p);
  f.q[1] = *(const u32x4*)(p + 16);
  return f.v;
}

// A fragment from LDS (padded stride HPAD).
__device__ __forceinline__ v16bf load_a_lds(const __bf16* S, int kbase, int lane) {
  int row = lane & 15;
  int kh  = (lane >> 4) * 8;
  const __bf16* p = S + row * HPAD + kbase + kh;
  FragU f;
  f.q[0] = *(const u32x4*)(p);
  f.q[1] = *(const u32x4*)(p + 16);
  return f.v;
}

// B fragment: packed contiguous 16 bf16 per lane per (nt,kt) tile.
__device__ __forceinline__ v16bf load_b_pk(const __bf16* __restrict__ Bp, int KT,
                                           int nt, int kt, int lane) {
  const __bf16* p = Bp + (((size_t)nt * KT + kt) * 32 + lane) * 16;
  FragU f;
  f.q[0] = *(const u32x4*)(p);
  f.q[1] = *(const u32x4*)(p + 8);
  return f.v;
}

// Async-stage one 16 x Hn bf16 tile (row-major, stride lda) into LDS with row
// stride HPAD, using GLOBAL_LOAD_ASYNC_TO_LDS_B128 (16B per issue).
// Caller must later execute async_wait() + __syncthreads() before reading.
__device__ __forceinline__ void async_stage_a(const __bf16* __restrict__ gsrc, int lda,
                                              int mbase, __bf16* sdst,
                                              int tid, int nthreads) {
  constexpr int CH = Hn * 2 / 16;               // 64 16B-chunks per row
  for (int i = tid; i < 16 * CH; i += nthreads) {
    int row = i >> 6;                           // /CH
    int ch  = i & (CH - 1);
    const __bf16* g = gsrc + (size_t)(mbase + row) * lda + ch * 8;
    unsigned int lds_off = (unsigned int)(uintptr_t)(sdst + row * HPAD + ch * 8);
    unsigned long long ga = (unsigned long long)(uintptr_t)g;
    asm volatile("global_load_async_to_lds_b128 %0, %1, off"
                 :: "v"(lds_off), "v"(ga)
                 : "memory");
  }
}

__device__ __forceinline__ void async_wait() {
  asm volatile("s_wait_asynccnt 0" ::: "memory");
}

// ---- K0: pack fp32 (K x N) weight into bf16 WMMA-B fragment order ---------
// Lane l: column c = l&15, half = l>>4 ; element e: K = kt*32 + (e>>3)*16 + half*8 + (e&7).
__global__ void pack_b(const float* __restrict__ W, __bf16* __restrict__ out,
                       int K, int N) {
  int idx = blockIdx.x * blockDim.x + threadIdx.x;
  int KT = K / 32, NT = N / 16;
  int total = KT * NT * 32;
  if (idx >= total) return;
  int lane = idx & 31;
  int tile = idx >> 5;
  int kt = tile % KT;
  int nt = tile / KT;
  int c = lane & 15, half = lane >> 4;
  __bf16* dst = out + (size_t)idx * 16;
#pragma unroll
  for (int e = 0; e < 16; ++e) {
    int k = kt * 32 + (e >> 3) * 16 + half * 8 + (e & 7);
    dst[e] = f2bf(W[(size_t)k * N + nt * 16 + c]);
  }
}

// ---- K1: elementwise scans per (b,h) ---------------------------------------
__global__ void grud_scan(const float* __restrict__ C, const float* __restrict__ t,
                          const int* __restrict__ mask,
                          const float* __restrict__ w_gx, const float* __restrict__ b_gx,
                          __bf16* __restrict__ delta_bf, __bf16* __restrict__ xhat_bf,
                          __bf16* __restrict__ mask_bf) {
  int idx = blockIdx.x * blockDim.x + threadIdx.x;   // b*H + h
  if (idx >= Bn * Hn) return;
  int b = idx / Hn, h = idx % Hn;
  // pass 1: x_mean = sum(m*x)/max(sum(m),1)  (x broadcast of C over batch)
  float sm = 0.f, smx = 0.f;
  for (int l = 0; l < Ln; ++l) {
    float m = (float)mask[((size_t)b * Ln + l) * Hn + h];
    sm  += m;
    smx += m * C[l * Hn + h];
  }
  float x_mean = smx / fmaxf(sm, 1.0f);
  float wg = w_gx[h], bg = b_gx[h];
  float x_last = x_mean, d_prev = 0.f, m_prev = 1.f, t_prev = 0.f;
  for (int l = 0; l < Ln; ++l) {
    size_t row = (size_t)b * Ln + l;
    float m    = (float)mask[row * Hn + h];
    float xi   = C[l * Hn + h];
    float tcur = t[b * Ln + l];
    float dtl  = (l == 0) ? 0.f : (tcur - t_prev);
    t_prev = tcur;
    float delta = dtl + (1.f - m_prev) * d_prev;
    float gx    = expf(-fmaxf(0.f, wg * delta + bg));
    float xh    = m * xi + (1.f - m) * (gx * x_last + (1.f - gx) * x_mean);
    delta_bf[row * Hn + h] = f2bf(delta);
    xhat_bf [row * Hn + h] = f2bf(xh);
    mask_bf [row * Hn + h] = f2bf(m);
    x_last = m * xi + (1.f - m) * x_last;
    d_prev = delta;
    m_prev = m;
  }
}

// ---- K2: gamma_h GEMM: (BL x H) @ (H x H), epilogue exp(-relu(.+bias)) -----
__global__ __launch_bounds__(256) void gemm_gamma(const __bf16* __restrict__ A,
                                                  const __bf16* __restrict__ Bp,
                                                  const float* __restrict__ bias,
                                                  __bf16* __restrict__ D) {
  __shared__ __bf16 sA[16 * HPAD];
  int lane = threadIdx.x & 31, wave = threadIdx.x >> 5;
  int mt = blockIdx.x;
  int nt = blockIdx.y * 8 + wave;
  constexpr int KT = Hn / 32;

  async_stage_a(A, Hn, mt * 16, sA, threadIdx.x, 256);
  async_wait();
  __syncthreads();

  v8f acc = {};
#pragma unroll 4
  for (int kt = 0; kt < KT; ++kt) {
    __builtin_prefetch(Bp + (((size_t)nt * KT + kt + 1) * 32 + lane) * 16, 0, 0);
    v16bf a = load_a_lds(sA, kt * 32, lane);
    v16bf b = load_b_pk(Bp, KT, nt, kt, lane);
    acc = WMMA_BF16(a, b, acc);
  }
  int c = lane & 15, hh = lane >> 4;
  int n = nt * 16 + c;
  float bv = bias[n];
#pragma unroll
  for (int v = 0; v < 8; ++v) {
    int m = mt * 16 + v + 8 * hh;
    D[(size_t)m * Hn + n] = f2bf(expf(-fmaxf(0.f, acc[v] + bv)));
  }
}

// ---- K3: pre GEMM: x_hat@Wx + mask@Wm + b -> fp32 (BL x 3H) ----------------
__global__ __launch_bounds__(256) void gemm_pre(const __bf16* __restrict__ Ax,
                                                const __bf16* __restrict__ Am,
                                                const __bf16* __restrict__ Bx,
                                                const __bf16* __restrict__ Bm,
                                                const float* __restrict__ bias,
                                                float* __restrict__ P) {
  __shared__ __bf16 sAx[16 * HPAD];
  __shared__ __bf16 sAm[16 * HPAD];
  int lane = threadIdx.x & 31, wave = threadIdx.x >> 5;
  int mt = blockIdx.x;
  int nt = blockIdx.y * 8 + wave;
  constexpr int KT = Hn / 32;

  async_stage_a(Ax, Hn, mt * 16, sAx, threadIdx.x, 256);
  async_stage_a(Am, Hn, mt * 16, sAm, threadIdx.x, 256);
  async_wait();
  __syncthreads();

  v8f acc = {};
#pragma unroll 2
  for (int kt = 0; kt < KT; ++kt) {
    __builtin_prefetch(Bx + (((size_t)nt * KT + kt + 1) * 32 + lane) * 16, 0, 0);
    v16bf a1 = load_a_lds(sAx, kt * 32, lane);
    v16bf b1 = load_b_pk(Bx, KT, nt, kt, lane);
    acc = WMMA_BF16(a1, b1, acc);
    v16bf a2 = load_a_lds(sAm, kt * 32, lane);
    v16bf b2 = load_b_pk(Bm, KT, nt, kt, lane);
    acc = WMMA_BF16(a2, b2, acc);
  }
  int c = lane & 15, hh = lane >> 4;
  int n = nt * 16 + c;
  float bv = bias[n];
#pragma unroll
  for (int v = 0; v < 8; ++v) {
    int m = mt * 16 + v + 8 * hh;
    P[(size_t)m * H3n + n] = acc[v] + bv;
  }
}

// ---- K4: sequential h recurrence; 8 WGs, each owns 16 batch rows -----------
__global__ __launch_bounds__(512) void grud_recur(const __bf16* __restrict__ gam_bf,
                                                  const float* __restrict__ P,
                                                  const __bf16* __restrict__ Whp,
                                                  __bf16* __restrict__ hseq_bf) {
  __shared__ float  h_f[16 * HPAD];   // fp32 hidden state
  __shared__ float  z_f[16 * 512];    // update gate
  __shared__ __bf16 h_b[16 * HPAD];   // bf16 copy of h (WMMA A)
  __shared__ __bf16 a_b[16 * HPAD];   // bf16 r*h      (WMMA A)

  int b0   = blockIdx.x * 16;
  int lane = threadIdx.x & 31, wave = threadIdx.x >> 5;
  int c = lane & 15, hh = lane >> 4;
  constexpr int KT = Hn / 32;         // 16 k-steps

  for (int i = threadIdx.x; i < 16 * HPAD; i += 512) h_f[i] = 0.f;
  __syncthreads();

  for (int l = 0; l < Ln; ++l) {
    // (1) h <- gamma_h * h ; refresh bf16 copy
    for (int i = threadIdx.x; i < 16 * Hn; i += 512) {
      int m = i >> 9, j = i & 511;
      size_t row = (size_t)(b0 + m) * Ln + l;
      float g  = (float)gam_bf[row * Hn + j];
      float hv = h_f[m * HPAD + j] * g;
      h_f[m * HPAD + j] = hv;
      h_b[m * HPAD + j] = f2bf(hv);
    }
    __syncthreads();

    // (2) zr = sigmoid(pre[:, :2H] + h @ Wh[:, :2H]) ; 64 n-tiles / 16 waves
#pragma unroll
    for (int q = 0; q < 4; ++q) {
      int nt = wave * 4 + q;                        // 0..63
      v8f acc = {};
#pragma unroll 4
      for (int kt = 0; kt < KT; ++kt) {
        __builtin_prefetch(Whp + (((size_t)nt * KT + kt + 1) * 32 + lane) * 16, 0, 0);
        v16bf a = load_a_lds(h_b, kt * 32, lane);
        v16bf b = load_b_pk(Whp, KT, nt, kt, lane);
        acc = WMMA_BF16(a, b, acc);
      }
      int n = nt * 16 + c;
#pragma unroll
      for (int v = 0; v < 8; ++v) {
        int m = v + 8 * hh;
        size_t prow = ((size_t)(b0 + m) * Ln + l) * H3n;
        float s = 1.f / (1.f + expf(-(acc[v] + P[prow + n])));
        if (n < Hn) {
          z_f[m * 512 + n] = s;                     // z
        } else {
          int j = n - Hn;                           // r -> r*h (next WMMA A)
          a_b[m * HPAD + j] = f2bf(s * h_f[m * HPAD + j]);
        }
      }
    }
    __syncthreads();

    // (3) h_tilde = tanh(pre[:, 2H:] + (r*h) @ Wh[:, 2H:]) ; h update; emit
#pragma unroll
    for (int q = 0; q < 2; ++q) {
      int nt = 64 + wave * 2 + q;                   // 64..95
      v8f acc = {};
#pragma unroll 4
      for (int kt = 0; kt < KT; ++kt) {
        v16bf a = load_a_lds(a_b, kt * 32, lane);
        v16bf b = load_b_pk(Whp, KT, nt, kt, lane);
        acc = WMMA_BF16(a, b, acc);
      }
      int n = nt * 16 + c;                          // 1024..1535
      int j = n - 1024;
#pragma unroll
      for (int v = 0; v < 8; ++v) {
        int m = v + 8 * hh;
        size_t prow = ((size_t)(b0 + m) * Ln + l) * H3n;
        float ht = tanhf(acc[v] + P[prow + n]);
        float z  = z_f[m * 512 + j];
        float hn = (1.f - z) * h_f[m * HPAD + j] + z * ht;
        h_f[m * HPAD + j] = hn;
        hseq_bf[((size_t)(b0 + m) * Ln + l) * Hn + j] = f2bf(hn);
      }
    }
    __syncthreads();
  }
}

// ---- K5: output GEMM: h_seq @ W_out + b_out -> fp32 ------------------------
__global__ __launch_bounds__(256) void gemm_out(const __bf16* __restrict__ A,
                                                const __bf16* __restrict__ Bp,
                                                const float* __restrict__ bias,
                                                float* __restrict__ Dout) {
  __shared__ __bf16 sA[16 * HPAD];
  int lane = threadIdx.x & 31, wave = threadIdx.x >> 5;
  int mt = blockIdx.x;
  int nt = blockIdx.y * 8 + wave;
  constexpr int KT = Hn / 32;

  async_stage_a(A, Hn, mt * 16, sA, threadIdx.x, 256);
  async_wait();
  __syncthreads();

  v8f acc = {};
#pragma unroll 4
  for (int kt = 0; kt < KT; ++kt) {
    __builtin_prefetch(Bp + (((size_t)nt * KT + kt + 1) * 32 + lane) * 16, 0, 0);
    v16bf a = load_a_lds(sA, kt * 32, lane);
    v16bf b = load_b_pk(Bp, KT, nt, kt, lane);
    acc = WMMA_BF16(a, b, acc);
  }
  int c = lane & 15, hh = lane >> 4;
  int n = nt * 16 + c;
  float bv = bias[n];
#pragma unroll
  for (int v = 0; v < 8; ++v) {
    int m = mt * 16 + v + 8 * hh;
    Dout[(size_t)m * OUTn + n] = acc[v] + bv;
  }
}

// ---------------------------------------------------------------------------
extern "C" void kernel_launch(void* const* d_in, const int* in_sizes, int n_in,
                              void* d_out, int out_size, void* d_ws, size_t ws_size,
                              hipStream_t stream) {
  const float* C     = (const float*)d_in[0];
  const float* t     = (const float*)d_in[1];
  const int*   mask  = (const int*)d_in[2];
  const float* Wx    = (const float*)d_in[3];
  const float* Wh    = (const float*)d_in[4];
  const float* Wm    = (const float*)d_in[5];
  const float* bias  = (const float*)d_in[6];
  const float* w_gx  = (const float*)d_in[7];
  const float* b_gx  = (const float*)d_in[8];
  const float* W_gh  = (const float*)d_in[9];
  const float* b_gh  = (const float*)d_in[10];
  const float* W_out = (const float*)d_in[11];
  const float* b_out = (const float*)d_in[12];
  (void)in_sizes; (void)n_in; (void)out_size; (void)ws_size;

  char* ws = (char*)d_ws;
  size_t off = 0;
  auto alloc = [&](size_t bytes) {
    char* p = ws + off;
    off += (bytes + 255) & ~(size_t)255;
    return p;
  };
  __bf16* Wx_p   = (__bf16*)alloc((size_t)Hn * H3n * 2);
  __bf16* Wm_p   = (__bf16*)alloc((size_t)Hn * H3n * 2);
  __bf16* Wh_p   = (__bf16*)alloc((size_t)Hn * H3n * 2);
  __bf16* Wgh_p  = (__bf16*)alloc((size_t)Hn * Hn * 2);
  __bf16* Wout_p = (__bf16*)alloc((size_t)Hn * OUTn * 2);
  __bf16* delta_bf = (__bf16*)alloc((size_t)BLn * Hn * 2);
  __bf16* xhat_bf  = (__bf16*)alloc((size_t)BLn * Hn * 2);
  __bf16* mask_bf  = (__bf16*)alloc((size_t)BLn * Hn * 2);
  __bf16* gam_bf   = (__bf16*)alloc((size_t)BLn * Hn * 2);
  __bf16* hseq_bf  = (__bf16*)alloc((size_t)BLn * Hn * 2);
  float*  P        = (float*)alloc((size_t)BLn * H3n * 4);

  auto packLaunch = [&](const float* W, __bf16* dst, int K, int N) {
    int total = (K / 32) * (N / 16) * 32;
    pack_b<<<(total + 255) / 256, 256, 0, stream>>>(W, dst, K, N);
  };
  packLaunch(Wx,    Wx_p,   Hn, H3n);
  packLaunch(Wm,    Wm_p,   Hn, H3n);
  packLaunch(Wh,    Wh_p,   Hn, H3n);
  packLaunch(W_gh,  Wgh_p,  Hn, Hn);
  packLaunch(W_out, Wout_p, Hn, OUTn);

  grud_scan<<<(Bn * Hn) / 256, 256, 0, stream>>>(C, t, mask, w_gx, b_gx,
                                                 delta_bf, xhat_bf, mask_bf);
  gemm_gamma<<<dim3(BLn / 16, (Hn / 16) / 8), 256, 0, stream>>>(delta_bf, Wgh_p,
                                                                b_gh, gam_bf);
  gemm_pre<<<dim3(BLn / 16, (H3n / 16) / 8), 256, 0, stream>>>(xhat_bf, mask_bf,
                                                               Wx_p, Wm_p, bias, P);
  grud_recur<<<Bn / 16, 512, 0, stream>>>(gam_bf, P, Wh_p, hseq_bf);
  gemm_out<<<dim3(BLn / 16, (OUTn / 16) / 8), 256, 0, stream>>>(hseq_bf, Wout_p,
                                                                b_out, (float*)d_out);
}
